// XaiYoloWrapper_790273982882
// MI455X (gfx1250) — matrix operand order repaired and checked by reading, not compiled, not run on hardware
//
#include <hip/hip_runtime.h>

// ---------------------------------------------------------------------------
// YOLO NMS post-processing for MI455X (gfx1250, wave32).
//   K1 scores  : 1 pass over 137MB prediction -> per-anchor sort key + cls id
//   K2 select  : per-image radix-select top-4096 threshold + gather + bitonic
//                sort of (key<<32 | ~idx) u64s in LDS; exports valid count
//   K3 nms     : per-image greedy NMS in LDS, serial loop bounded by valid
//                count with MAX_DET early-exit; emits keep*obj weights (f16)
//   K4 gather  : transposed f16 class buffer clsT[b][c][k] (TDM-friendly 2D)
//   K5 reduce  : out[b,c] = sum_k w_k*cls[k,c] via v_wmma_f32_16x16x32_f16,
//                B tiles staged into LDS by the Tensor Data Mover (double-
//                buffered, s_wait_tensorcnt); global-load fallback otherwise.
// ---------------------------------------------------------------------------

typedef __attribute__((ext_vector_type(16))) _Float16 v16h;
typedef __attribute__((ext_vector_type(8)))  _Float16 v8h;
typedef __attribute__((ext_vector_type(8)))  float    v8f;
typedef __attribute__((ext_vector_type(4)))  unsigned u32x4;
typedef __attribute__((ext_vector_type(8)))  int      i32x8;
typedef __attribute__((ext_vector_type(4)))  int      i32x4;

#define B_      16
#define N_      25200
#define NC_     80
#define D_      85
#define K_      4096
#define CAP_    8192
#define CH_     256        // candidates per TDM tile (per 16-class row block)
#define NCH_    (K_ / CH_)
#define CONF_   0.25f
#define IOU_T_  0.45f
#define MAX_WH_ 7680.0f
#define MAX_DET_ 1000

// TDM dispatch across probe-documented toolchain variants:
//   _d2 builtin (2 descriptor groups) -> 3 args, ideal for 2D tensors
//   header present (amdgpu-toolchain clang-23) -> 6-arg tensor_load_to_lds
//   otherwise (ROCm 7.2 clang-22) -> 5-arg tensor_load_to_lds
#if defined(__has_builtin)
#  if __has_builtin(__builtin_amdgcn_tensor_load_to_lds_d2)
#    define TDM_MODE 3
#  elif __has_builtin(__builtin_amdgcn_tensor_load_to_lds)
#    if __has_include(<hip/amd_detail/amd_gfx1250_TDM.h>)
#      define TDM_MODE 6
#    else
#      define TDM_MODE 5
#    endif
#  endif
#endif
#ifndef TDM_MODE
#  define TDM_MODE 0
#endif

__device__ __forceinline__ void wait_tensorcnt0() {
#if __has_builtin(__builtin_amdgcn_s_wait_tensorcnt)
  __builtin_amdgcn_s_wait_tensorcnt((short)0);
#else
  asm volatile("s_wait_tensorcnt 0x0" ::: "memory");
#endif
}
__device__ __forceinline__ void wait_tensorcnt1() {
#if __has_builtin(__builtin_amdgcn_s_wait_tensorcnt)
  __builtin_amdgcn_s_wait_tensorcnt((short)1);
#else
  asm volatile("s_wait_tensorcnt 0x1" ::: "memory");
#endif
}
__device__ __forceinline__ void wait_dscnt0() {
  asm volatile("s_wait_dscnt 0x0" ::: "memory");
}

#if TDM_MODE
// Issue one TDM 2D tile load: 16 rows x CH_ halfs, row stride K_ halfs,
// packed contiguously into LDS at ldsOff. D# per ISA 08_async_tensor 8.3/8.4.
__device__ __forceinline__ void tdm_load_tile(unsigned ldsOff, const void* gp) {
  unsigned long long ga = (unsigned long long)gp;
  u32x4 g0;
  g0[0] = 1u;                                    // count=1, user mode
  g0[1] = ldsOff;                                // lds_addr [63:32]
  g0[2] = (unsigned)(ga & 0xFFFFFFFFull);        // global_addr lo
  g0[3] = (unsigned)((ga >> 32) & 0x01FFFFFFull) // global_addr hi (57b total)
        | (2u << 30);                            // type = 2 ("image")
  i32x8 g1;
  g1[0] = (int)(1u << 16);           // wg_mask=0, data_size=1 (2 bytes)
  g1[1] = (int)((unsigned)CH_ << 16);// [47:32]=atomic_addr=0, tensor_dim0 lo16
  g1[2] = (int)(16u << 16);          // tensor_dim0 hi16=0, tensor_dim1 lo16=16
  g1[3] = (int)((unsigned)CH_ << 16);// tensor_dim1 hi16=0, tile_dim0=CH_
  g1[4] = 16;                        // tile_dim1=16, tile_dim2=0
  g1[5] = (int)K_;                   // tensor_dim0_stride lo32 = 4096
  g1[6] = 0;                         // stride hi16=0, dim1_stride lo16=0
  g1[7] = 0;
#if TDM_MODE == 3
  __builtin_amdgcn_tensor_load_to_lds_d2(g0, g1, 0);
#elif TDM_MODE == 5
  i32x4 z4 = {};
  __builtin_amdgcn_tensor_load_to_lds(g0, g1, z4, z4, 0);
#else  // TDM_MODE == 6
  i32x4 z4 = {};
  i32x8 z8 = {};
  __builtin_amdgcn_tensor_load_to_lds(g0, g1, z4, z4, z8, 0);
#endif
}
#endif

// -------------------- K1: per-anchor score key + class id ------------------
__global__ void k_scores(const float* __restrict__ pred,
                         unsigned* __restrict__ keys,
                         unsigned* __restrict__ clsArr) {
  int id = blockIdx.x * blockDim.x + threadIdx.x;
  if (id >= B_ * N_) return;
  const float* p = pred + (size_t)id * D_;
  float obj = p[4];
  float best = -1.0f;
  int bc = 0;
  for (int c = 0; c < NC_; ++c) {
    float v = p[5 + c] * obj;
    if (v > best) { best = v; bc = c; }   // first-max => matches jnp.argmax
  }
  bool valid = (obj > CONF_) && (best > CONF_);
  keys[id] = valid ? (__float_as_uint(best) | 0x80000000u) : 0u;
  clsArr[id] = (unsigned)bc;
}

// -------------------- K2: per-image top-K select + stable sort -------------
__global__ void k_select(const unsigned* __restrict__ keys,
                         unsigned* __restrict__ candKey,
                         unsigned* __restrict__ candIdx,
                         unsigned* __restrict__ candCnt) {
  extern __shared__ unsigned char smemRaw[];
  unsigned long long* buf = (unsigned long long*)smemRaw;
  unsigned* hist = (unsigned*)(smemRaw + (size_t)CAP_ * 8);
  unsigned* sh   = hist + 256;   // sh[0]=digit sh[1]=remaining sh[2]=count
  const int b = blockIdx.x;
  const unsigned* kb = keys + (size_t)b * N_;
  const int tid = threadIdx.x, nthr = blockDim.x;

  unsigned prefix = 0, remaining = K_;
  for (int shift = 24; shift >= 0; shift -= 8) {
    for (int i = tid; i < 256; i += nthr) hist[i] = 0;
    __syncthreads();
    const unsigned hmask = (shift == 24) ? 0u : (0xFFFFFFFFu << (shift + 8));
    for (int a = tid; a < N_; a += nthr) {
      unsigned k = kb[a];
      if (((k ^ prefix) & hmask) == 0u)
        atomicAdd(&hist[(k >> shift) & 0xFFu], 1u);
    }
    __syncthreads();
    if (tid == 0) {
      unsigned cum = 0, digit = 0, rem = remaining;
      for (int v = 255; v >= 0; --v) {
        unsigned c = hist[v];
        if (rem <= cum + c) { digit = (unsigned)v; rem -= cum; break; }
        cum += c;
      }
      sh[0] = digit; sh[1] = rem;
    }
    __syncthreads();
    prefix |= sh[0] << shift;
    remaining = sh[1];
    __syncthreads();
  }
  const unsigned T = prefix;

  if (tid == 0) sh[2] = 0;
  __syncthreads();
  for (int a = tid; a < N_; a += nthr) {
    unsigned k = kb[a];
    bool take = (T > 0u) ? (k >= T) : (k > 0u);
    if (take) {
      unsigned pos = atomicAdd(&sh[2], 1u);
      if (pos < CAP_)
        buf[pos] = ((unsigned long long)k << 32) | (unsigned)(~(unsigned)a);
    }
  }
  __syncthreads();
  unsigned cnt = sh[2]; if (cnt > CAP_) cnt = CAP_;
  if (tid == 0) candCnt[b] = (cnt < K_) ? cnt : (unsigned)K_;
  for (int i = tid; i < CAP_; i += nthr)
    if (i >= (int)cnt) buf[i] = 0ull;
  __syncthreads();

  // bitonic sort, descending by (key, ~idx) => stable top_k order
  for (unsigned kk = 2; kk <= CAP_; kk <<= 1) {
    for (unsigned j = kk >> 1; j > 0; j >>= 1) {
      for (unsigned i = (unsigned)tid; i < CAP_; i += (unsigned)nthr) {
        unsigned ixj = i ^ j;
        if (ixj > i) {
          unsigned long long a0 = buf[i], a1 = buf[ixj];
          bool descBlock = ((i & kk) == 0u);
          if (descBlock == (a0 < a1)) { buf[i] = a1; buf[ixj] = a0; }
        }
      }
      __syncthreads();
    }
  }

  for (int r = tid; r < K_; r += nthr) {
    unsigned long long v = buf[r];
    candKey[b * K_ + r] = (unsigned)(v >> 32);
    candIdx[b * K_ + r] = ~(unsigned)(v & 0xFFFFFFFFull);
  }
}

// -------------------- K3: per-image greedy NMS in LDS ----------------------
__global__ void k_nms(const float* __restrict__ pred,
                      const unsigned* __restrict__ clsArr,
                      const unsigned* __restrict__ candKey,
                      const unsigned* __restrict__ candIdx,
                      const unsigned* __restrict__ candCnt,
                      _Float16* __restrict__ wOut) {
  extern __shared__ unsigned char smemRaw[];
  float* x1 = (float*)smemRaw;
  float* y1 = x1 + K_; float* x2 = y1 + K_; float* y2 = x2 + K_;
  float* ar = y2 + K_; float* ob = ar + K_;
  unsigned* sup = (unsigned*)(ob + K_);
  unsigned* msc = sup + K_;        // [0],[1]=keep double-buffer, [2]=count
  const int b = blockIdx.x, tid = threadIdx.x, nthr = blockDim.x;
  const int L = (int)candCnt[b];   // valid candidates (sorted first)

  for (int r = tid; r < L; r += nthr) {
    unsigned idx = candIdx[b * K_ + r];
    const float* p = pred + ((size_t)b * N_ + idx) * D_;
    float cx = p[0], cy = p[1], w = p[2], h = p[3];
    float off = (float)clsArr[(size_t)b * N_ + idx] * MAX_WH_;
    float lx = cx - w * 0.5f + off, ly = cy - h * 0.5f + off;
    float hx = cx + w * 0.5f + off, hy = cy + h * 0.5f + off;
    x1[r] = lx; y1[r] = ly; x2[r] = hx; y2[r] = hy;
    ar[r] = (hx - lx) * (hy - ly);   // > 0 (w,h >= 1)
    ob[r] = p[4];
    sup[r] = 0u;
  }
  if (tid == 0) msc[2] = 0u;
  __syncthreads();

  int r = 0;
  for (; r < L; ++r) {
    if (tid == 0) {
      unsigned cnt = msc[2];
      unsigned k = (sup[r] == 0u) && (cnt < MAX_DET_);
      msc[r & 1] = k;
      if (k) msc[2] = cnt + 1u;
      wOut[(size_t)b * K_ + r] = k ? (_Float16)ob[r] : (_Float16)0.0f;
    }
    __syncthreads();
    if (msc[2] >= MAX_DET_ && !msc[r & 1]) { ++r; break; }  // uniform
    if (msc[r & 1]) {                                       // uniform
      float rx1 = x1[r], ry1 = y1[r], rx2 = x2[r], ry2 = y2[r], rar = ar[r];
      for (int j = tid; j < L; j += nthr) {
        float lx = fmaxf(rx1, x1[j]), ly = fmaxf(ry1, y1[j]);
        float hx = fminf(rx2, x2[j]), hy = fminf(ry2, y2[j]);
        float iw = fmaxf(hx - lx, 0.0f), ih = fmaxf(hy - ly, 0.0f);
        float inter = iw * ih;
        float iou = inter / (rar + ar[j] - inter);
        if (iou > IOU_T_) sup[j] = 1u;   // includes j==r (matches reference)
      }
      __syncthreads();
    }
  }
  // disjoint-range zero fill for the tail (no same-address write hazard)
  for (int t = r + tid; t < K_; t += nthr)
    wOut[(size_t)b * K_ + t] = (_Float16)0.0f;
}

// -------------------- K4: gather transposed f16 class buffer ---------------
__global__ void k_gather(const float* __restrict__ pred,
                         const unsigned* __restrict__ candKey,
                         const unsigned* __restrict__ candIdx,
                         _Float16* __restrict__ clsT) {
  long long t = (long long)blockIdx.x * blockDim.x + threadIdx.x;
  if (t >= (long long)B_ * NC_ * K_) return;
  int r = (int)(t % K_);
  int c = (int)((t / K_) % NC_);
  int b = (int)(t / ((long long)K_ * NC_));
  unsigned key = candKey[b * K_ + r];
  _Float16 v = (_Float16)0.0f;
  if (key > 0u) {
    unsigned idx = candIdx[b * K_ + r];
    v = (_Float16)pred[((size_t)b * N_ + idx) * D_ + 5 + c];
  }
  clsT[((size_t)b * NC_ + c) * K_ + r] = v;
}

// -------------------- K5: WMMA weighted reduction --------------------------
// One wave per (image, 16-class tile). A row 0 = keep weights, built
// branch-free: every lane loads the slice its half-wave needs and multiplies
// by a 0/1 mask (only lanes 0 and 16 carry data per the ISA A layout).
// B here (contiguous k, as A): lanes 0-15 <-> K 0-15, lanes 16-31 <-> K 16-31, col = lane%16.
__global__ void __launch_bounds__(32) k_reduce(const _Float16* __restrict__ clsT,
                                               const _Float16* __restrict__ w,
                                               float* __restrict__ out) {
  const int b    = blockIdx.x / 5;
  const int c0   = (blockIdx.x % 5) * 16;
  const int lane = threadIdx.x;
  const _Float16* wb = w + (size_t)b * K_;
  const int n    = lane & 15;
  const int aoff = (lane & 16) ? 16 : 0;           // lane16 half-wave slice
  const int boff = (lane & 16) ? 16 : 0;
  const _Float16 m = (lane == 0 || lane == 16) ? (_Float16)1.0f : (_Float16)0.0f;

  v8f acc = {};
#if TDM_MODE
  __shared__ _Float16 ldsB[2][16 * CH_];           // 16KB double buffer
  const _Float16* gbase = clsT + ((size_t)b * NC_ + c0) * K_;
  tdm_load_tile((unsigned)(unsigned long long)(const void*)&ldsB[0][0], gbase);
  for (int c = 0; c < NCH_; ++c) {
    if (c + 1 < NCH_) {
      // DS pipe and TDM engine are unordered: retire our in-flight ds_loads
      // on the target buffer before the async engine may overwrite it.
      wait_dscnt0();
      tdm_load_tile((unsigned)(unsigned long long)(const void*)&ldsB[(c + 1) & 1][0],
                    gbase + (size_t)(c + 1) * CH_);
      wait_tensorcnt1();                           // current tile landed
    } else {
      wait_tensorcnt0();
    }
    const _Float16* lb = &ldsB[c & 1][0] + n * CH_ + boff;
    for (int s = 0; s < CH_ / 32; ++s) {
      const int k0 = c * CH_ + s * 32;
      v8h lo = *(const v8h*)(wb + k0 + aoff);
      v8h hi = *(const v8h*)(wb + k0 + aoff + 8);
      v16h a;
      for (int j = 0; j < 8; ++j) { a[j] = lo[j] * m; a[8 + j] = hi[j] * m; }
      v16h bm = *(const v16h*)(lb + s * 32);       // ds_load from staged tile
      acc = __builtin_amdgcn_wmma_f32_16x16x32_f16(false, a, false, bm,
                                                   (short)0, acc, false, false);
    }
  }
#else
  const _Float16* brow = clsT + ((size_t)b * NC_ + c0 + n) * K_ + boff;
  for (int k0 = 0; k0 < K_; k0 += 32) {
    __builtin_prefetch(brow + k0 + 512, 0, 1);     // global_prefetch_b8
    v8h lo = *(const v8h*)(wb + k0 + aoff);
    v8h hi = *(const v8h*)(wb + k0 + aoff + 8);
    v16h a;
    for (int j = 0; j < 8; ++j) { a[j] = lo[j] * m; a[8 + j] = hi[j] * m; }
    v16h bm = *(const v16h*)(brow + k0);
    acc = __builtin_amdgcn_wmma_f32_16x16x32_f16(false, a, false, bm,
                                                 (short)0, acc, false, false);
  }
#endif
  // D row 0 lives in VGPR0 of lanes 0..15
  if (lane < 16) out[b * NC_ + c0 + lane] = acc[0];
}

// ---------------------------------------------------------------------------
extern "C" void kernel_launch(void* const* d_in, const int* in_sizes, int n_in,
                              void* d_out, int out_size, void* d_ws, size_t ws_size,
                              hipStream_t stream) {
  (void)in_sizes; (void)n_in; (void)out_size; (void)ws_size;
  const float* pred = (const float*)d_in[0];
  float* out = (float*)d_out;

  unsigned char* ws = (unsigned char*)d_ws;
  size_t off = 0;
  auto alloc = [&](size_t bytes) -> void* {
    void* p = ws + off;
    off += (bytes + 255) & ~(size_t)255;
    return p;
  };
  unsigned* keys    = (unsigned*)alloc((size_t)B_ * N_ * 4);       // 1.6MB
  unsigned* clsArr  = (unsigned*)alloc((size_t)B_ * N_ * 4);       // 1.6MB
  unsigned* candKey = (unsigned*)alloc((size_t)B_ * K_ * 4);       // 256KB
  unsigned* candIdx = (unsigned*)alloc((size_t)B_ * K_ * 4);       // 256KB
  unsigned* candCnt = (unsigned*)alloc((size_t)B_ * 4);
  _Float16* wBuf    = (_Float16*)alloc((size_t)B_ * K_ * 2);       // 128KB
  _Float16* clsT    = (_Float16*)alloc((size_t)B_ * NC_ * K_ * 2); // 10.5MB

  k_scores<<<(B_ * N_ + 255) / 256, 256, 0, stream>>>(pred, keys, clsArr);

  size_t selLds = (size_t)CAP_ * 8 + 256 * 4 + 16;                 // ~65.5KB
  k_select<<<B_, 1024, selLds, stream>>>(keys, candKey, candIdx, candCnt);

  size_t nmsLds = (size_t)6 * K_ * 4 + (size_t)K_ * 4 + 16;        // ~112KB
  k_nms<<<B_, 1024, nmsLds, stream>>>(pred, clsArr, candKey, candIdx, candCnt,
                                      wBuf);

  k_gather<<<(B_ * NC_ * K_ + 255) / 256, 256, 0, stream>>>(pred, candKey,
                                                            candIdx, clsT);

  k_reduce<<<B_ * 5, 32, 0, stream>>>(clsT, wBuf, out);
}
